// GConv2d_7232724926789
// MI455X (gfx1250) — matrix operand order repaired and checked by reading, not compiled, not run on hardware
//
#include <hip/hip_runtime.h>
#include <hip/hip_bf16.h>

typedef __attribute__((ext_vector_type(8)))  float  v8f;
typedef __attribute__((ext_vector_type(8)))  __bf16 bf16x8;
typedef __attribute__((ext_vector_type(16))) __bf16 bf16x16;

// ---------------- problem constants ----------------
#define B_    16
#define CIN   256
#define HW    4096     // 64*64
#define OCH   256      // 4 orientations * 64
#define MBLK  128
#define NBLK  128      // 2 full image rows
#define SROWS 264      // slab: 4 image rows x 66 padded cols (x = -1..64)
#define SSTR  72       // 64 channels + 8 pad (144B rows, 16B-aligned, bank-spread)
#define WPS   147456   // per-k-slice weight count: 64*256*9

__device__ __forceinline__ unsigned short f2bf(float f) {
    unsigned int u = __float_as_uint(f);
    unsigned int r = u + 0x7fffu + ((u >> 16) & 1u);   // round-nearest-even
    return (unsigned short)(r >> 16);
}

#if __has_builtin(__builtin_amdgcn_cvt_pk_bf16_f32)
__device__ __forceinline__ unsigned int pk_bf16(float a, float b) {
    typedef __attribute__((ext_vector_type(2))) __bf16 bf16x2;
    bf16x2 v = __builtin_amdgcn_cvt_pk_bf16_f32(a, b);   // a -> [15:0], b -> [31:16]
    unsigned int u;
    __builtin_memcpy(&u, &v, 4);
    return u;
}
#else
__device__ __forceinline__ unsigned int pk_bf16(float a, float b) {
    return (unsigned int)f2bf(a) | ((unsigned int)f2bf(b) << 16);
}
#endif

// ---- 1) global average pool: one block per (b,c) ----
__global__ __launch_bounds__(256) void pool_kernel(const float* __restrict__ x,
                                                   float* __restrict__ pooled) {
    __shared__ float red[256];
    int bc = blockIdx.x;
    int t  = threadIdx.x;
    const float* p = x + (size_t)bc * HW;
    float s = 0.f;
    #pragma unroll
    for (int i = 0; i < 16; ++i) s += p[t + i * 256];
    red[t] = s;
    __syncthreads();
    for (int off = 128; off > 0; off >>= 1) {
        if (t < off) red[t] += red[t + off];
        __syncthreads();
    }
    if (t == 0) pooled[bc] = red[0] * (1.f / 4096.f);
}

// ---- 2) routing: logits + softmax over K=4 ----
__global__ __launch_bounds__(64) void route_kernel(const float* __restrict__ pooled,
                                                   const float* __restrict__ Wfc,
                                                   const float* __restrict__ bfc,
                                                   float* __restrict__ coeffs) {
    __shared__ float lg[64];
    int t = threadIdx.x;
    {
        int b = t >> 2, k = t & 3;
        float s = bfc[k];
        const float* pr = pooled + b * CIN;
        const float* wr = Wfc + k * CIN;
        for (int c = 0; c < CIN; ++c) s += pr[c] * wr[c];
        lg[t] = s;
    }
    __syncthreads();
    if (t < 16) {
        float a0 = lg[t*4+0], a1 = lg[t*4+1], a2 = lg[t*4+2], a3 = lg[t*4+3];
        float m  = fmaxf(fmaxf(a0, a1), fmaxf(a2, a3));
        float e0 = __expf(a0 - m), e1 = __expf(a1 - m);
        float e2 = __expf(a2 - m), e3 = __expf(a3 - m);
        float inv = 1.f / (e0 + e1 + e2 + e3);
        coeffs[t*4+0] = e0 * inv;
        coeffs[t*4+1] = e1 * inv;
        coeffs[t*4+2] = e2 * inv;
        coeffs[t*4+3] = e3 * inv;
    }
}

// ---- 3) mix bank -> per-sample weights, all 4 rotations pre-materialized ----
// Output layout: wbf4[b][tap][o][c] (bf16), o = r*64+oc, tap of the ORIENTED filter.
__global__ __launch_bounds__(256) void mix_kernel(const float* __restrict__ coeffs,
                                                  const float* __restrict__ bank,
                                                  unsigned short* __restrict__ wbf4) {
    int e   = blockIdx.x * 256 + threadIdx.x;    // over 16*64*256*9
    int b   = e / WPS;
    int rem = e - b * WPS;                        // oc*2304 + c*9 + tap
    int oc   = rem / 2304;
    int rem2 = rem - oc * 2304;
    int c    = rem2 / 9;
    int tap  = rem2 - c * 9;
    int bi = tap / 3, bj = tap - bi * 3;

    float v = coeffs[b*4+0] * bank[0*WPS + rem]
            + coeffs[b*4+1] * bank[1*WPS + rem]
            + coeffs[b*4+2] * bank[2*WPS + rem]
            + coeffs[b*4+3] * bank[3*WPS + rem];
    unsigned short h = f2bf(v);

    // where base tap (bi,bj) lands in each rotated (oriented) filter
    int t0 = bi * 3 + bj;                 // r=0
    int t1 = (2 - bj) * 3 + bi;           // r=1
    int t2 = (2 - bi) * 3 + (2 - bj);     // r=2
    int t3 = bj * 3 + (2 - bi);           // r=3

    size_t bb = (size_t)b * 9;
    wbf4[((bb + t0) << 16) + (((0 << 6) + oc) << 8) + c] = h;
    wbf4[((bb + t1) << 16) + (((1 << 6) + oc) << 8) + c] = h;
    wbf4[((bb + t2) << 16) + (((2 << 6) + oc) << 8) + c] = h;
    wbf4[((bb + t3) << 16) + (((3 << 6) + oc) << 8) + c] = h;
}

// ---- 4) shift+GEMM conv, bf16 WMMA ----
// Block tile M=128 x N=128. x staged once per 64-channel group into a zero-padded
// LDS pixel slab; the 9 taps become pure LDS pointer shifts (no loads/cvt/predicates).
__global__ __launch_bounds__(256) void gconv_wmma(const float* __restrict__ x,
                                                  const unsigned short* __restrict__ wbf4,
                                                  float* __restrict__ out) {
    __shared__ unsigned short As[MBLK * SSTR];    // weights [row][cc 0..63]
    __shared__ unsigned short Xs[SROWS * SSTR];   // padded pixel slab [sp][cc 0..63]

    int blk   = blockIdx.x;
    int b     = blk >> 6;            // 64 blocks/sample (2 mtiles * 32 ntiles)
    int rem   = blk & 63;
    int mtile = rem >> 5;
    int ntile = rem & 31;
    int Nbase = ntile * NBLK;
    int Mbase = mtile * MBLK;
    int y0    = ntile * 2;           // first image row of this N tile

    int tid  = threadIdx.x;
    int lane = tid & 31;
    int wave = tid >> 5;
    int wm   = wave & 3;             // M 32-row strip
    int wn   = wave >> 2;            // N half == image row parity within tile

    v8f acc[2][4] = {};

    // ---- A staging coords: 32 contiguous halfs per thread (4x b128) ----
    int arow = tid >> 1;                 // 0..127
    int ahalf = (tid & 1) << 5;          // 0 or 32 channel offset
    const unsigned short* wrowA = wbf4 + (((size_t)b * 9) << 16)
                                + ((size_t)(Mbase + arow) << 8) + ahalf;
    unsigned short* AsDst = &As[arow * SSTR + ahalf];

    const float* xs = x + ((size_t)b << 20);   // b*256*4096

    // ---- fragment coords (ISA 16-bit layouts), hoisted ----
    int aoff = (lane < 16) ? 0 : 8;
    const unsigned short* ApL[2];
    #pragma unroll
    for (int ms = 0; ms < 2; ++ms)
        ApL[ms] = &As[(wm * 32 + ms * 16 + (lane & 15)) * SSTR + aoff];
    int boff = (lane < 16) ? 0 : 8;
    int bx[4];                            // x coordinate of this lane's B column
    #pragma unroll
    for (int s = 0; s < 4; ++s) bx[s] = s * 16 + (lane & 15);

    for (int ch = 0; ch < 256; ch += 64) {
        // ---- stage zero-padded pixel slab (once per channel group) ----
        for (int sp = tid; sp < SROWS; sp += 256) {
            int iy = sp / 66;
            int sx = sp - iy * 66 - 1;           // -1..64
            int yr = y0 - 1 + iy;
            bool valid = ((unsigned)yr < 64u) & ((unsigned)sx < 64u);
            unsigned short* dst = &Xs[sp * SSTR];
            if (valid) {
                const float* xp = xs + ((size_t)ch << 12) + (yr << 6) + sx;
                #pragma unroll
                for (int g = 0; g < 8; ++g) {
                    uint4 w;
                    w.x = pk_bf16(xp[(size_t)(8*g+0) << 12], xp[(size_t)(8*g+1) << 12]);
                    w.y = pk_bf16(xp[(size_t)(8*g+2) << 12], xp[(size_t)(8*g+3) << 12]);
                    w.z = pk_bf16(xp[(size_t)(8*g+4) << 12], xp[(size_t)(8*g+5) << 12]);
                    w.w = pk_bf16(xp[(size_t)(8*g+6) << 12], xp[(size_t)(8*g+7) << 12]);
                    ((uint4*)dst)[g] = w;
                }
            } else {
                uint4 z = make_uint4(0u, 0u, 0u, 0u);
                #pragma unroll
                for (int g = 0; g < 8; ++g) ((uint4*)dst)[g] = z;
            }
        }

        for (int tap = 0; tap < 9; ++tap) {
            int ti = tap / 3;
            int dy = ti - 1;
            int dx = (tap - ti * 3) - 1;

            // stage A: 128 rows x 64 ch chunk, 4x b128 per thread
            const unsigned short* wt = wrowA + ((size_t)tap << 16) + ch;
            uint4 a0 = ((const uint4*)wt)[0];
            uint4 a1 = ((const uint4*)wt)[1];
            uint4 a2 = ((const uint4*)wt)[2];
            uint4 a3 = ((const uint4*)wt)[3];
            ((uint4*)AsDst)[0] = a0;
            ((uint4*)AsDst)[1] = a1;
            ((uint4*)AsDst)[2] = a2;
            ((uint4*)AsDst)[3] = a3;
            __syncthreads();   // covers slab (first tap) + As

            // B row pointers: tap = pure slab pointer shift, padding pre-baked
            const unsigned short* BpT[4];
            #pragma unroll
            for (int s = 0; s < 4; ++s) {
                int sp = (wn + dy + 1) * 66 + bx[s] + dx + 1;
                BpT[s] = &Xs[sp * SSTR + boff];
            }

            // ---- 16 WMMAs per wave between barriers ----
            #pragma unroll
            for (int cb = 0; cb < 2; ++cb) {
                int co = cb << 5;
                bf16x16 afrag[2];
                #pragma unroll
                for (int ms = 0; ms < 2; ++ms) {
                    bf16x8 alo = *(const bf16x8*)(ApL[ms] + co);
                    bf16x8 ahi = *(const bf16x8*)(ApL[ms] + co + 16);
                    afrag[ms] = __builtin_shufflevector(alo, ahi,
                        0,1,2,3,4,5,6,7,8,9,10,11,12,13,14,15);
                }
                #pragma unroll
                for (int s = 0; s < 4; ++s) {
                    bf16x8 blo = *(const bf16x8*)(BpT[s] + co);
                    bf16x8 bhi = *(const bf16x8*)(BpT[s] + co + 16);
                    bf16x16 bfrag = __builtin_shufflevector(blo, bhi,
                        0,1,2,3,4,5,6,7,8,9,10,11,12,13,14,15);
                    #pragma unroll
                    for (int ms = 0; ms < 2; ++ms) {
                        acc[ms][s] = __builtin_amdgcn_wmma_f32_16x16x32_bf16(
                            false, afrag[ms], false, bfrag, (short)0, acc[ms][s], false, false);
                    }
                }
            }
            __syncthreads();   // before next As / slab overwrite
        }
    }

    // ---- epilogue: C/D layout -> out[b][o][y][x] ----
    int ncol = lane & 15;
    int m0   = (lane < 16) ? 0 : 8;
    size_t ob = (size_t)b << 20;
    #pragma unroll
    for (int ms = 0; ms < 2; ++ms) {
        int obase = Mbase + wm * 32 + ms * 16;
        #pragma unroll
        for (int s = 0; s < 4; ++s) {
            int pcol = Nbase + wn * 64 + s * 16 + ncol;
            #pragma unroll
            for (int v = 0; v < 8; ++v) {
                int o = obase + m0 + v;
                out[ob + ((size_t)o << 12) + pcol] = acc[ms][s][v];
            }
        }
    }
}

extern "C" void kernel_launch(void* const* d_in, const int* in_sizes, int n_in,
                              void* d_out, int out_size, void* d_ws, size_t ws_size,
                              hipStream_t stream) {
    const float* x    = (const float*)d_in[0];   // [16,256,64,64]
    const float* Wfc  = (const float*)d_in[1];   // [4,256]
    const float* bfc  = (const float*)d_in[2];   // [4]
    const float* bank = (const float*)d_in[3];   // [4,64,256,3,3]
    float* out = (float*)d_out;                  // [16,256,64,64]

    // workspace layout
    float* pooled = (float*)d_ws;                                   // 4096 f32
    float* coeffs = pooled + 4096;                                  // 64 f32
    unsigned short* wbf4 = (unsigned short*)((char*)d_ws + 32768);  // 16*9*256*256 bf16 (~18.9MB)

    pool_kernel <<<B_ * CIN, 256, 0, stream>>>(x, pooled);
    route_kernel<<<1, 64, 0, stream>>>(pooled, Wfc, bfc, coeffs);
    mix_kernel  <<<(B_ * WPS) / 256, 256, 0, stream>>>(coeffs, bank, wbf4);
    gconv_wmma  <<<B_ * 2 * 32, 256, 0, stream>>>(x, wbf4, out);
}
